// NeuralMesh_28003186770312
// MI455X (gfx1250) — matrix operand order repaired and checked by reading, not compiled, not run on hardware
//
#include <hip/hip_runtime.h>
#include <hip/hip_bf16.h>
#include <math.h>

typedef __attribute__((ext_vector_type(16))) _Float16 v16h;
typedef __attribute__((ext_vector_type(8)))  float    v8f;

#define B_  256
#define K_  100
#define V_  1024
#define C_  128
#define KT  7      // ceil(100/16) k-tiles of 16 rows
#define BC  8      // 256 batches / 32 (WMMA K-dim)

// ---------------------------------------------------------------------------
// Kernel 1: build one-hot A operands in CDNA5 16-bit A-matrix (16x32) layout.
// Contiguous K (same as B): lanes 0-15 -> M=0..15 with K = j;
//                 lanes 16-31 -> M=0..15 with K = j+16.
// Stored as v16h per (ktile, bchunk, lane): 32 contiguous bytes per lane.
// ---------------------------------------------------------------------------
__global__ void NM_build_a(const int* __restrict__ label, v16h* __restrict__ Aop) {
    const int kt   = blockIdx.x / BC;
    const int bc   = blockIdx.x % BC;
    const int lane = threadIdx.x;          // 32 threads
    const int m    = lane & 15;
    const int base = (lane < 16) ? 0 : 16;
    v16h a;
#pragma unroll
    for (int j = 0; j < 16; ++j) {
        const int kdim = base + j;  // 0..31
        const int b    = bc * 32 + kdim;
        a[j] = (label[b] == kt * 16 + m) ? (_Float16)1.0f : (_Float16)0.0f;
    }
    Aop[(kt * BC + bc) * 32 + lane] = a;
}

// ---------------------------------------------------------------------------
// Kernel 2: denom[k,v] = segment_sum(visible) via LDS atomics (one block per v)
// ---------------------------------------------------------------------------
__global__ void NM_denom(const float* __restrict__ visible,
                         const int* __restrict__ label,
                         float* __restrict__ denom) {
    __shared__ float s[K_];
    const int v = blockIdx.x;
    for (int k = threadIdx.x; k < K_; k += blockDim.x) s[k] = 0.0f;
    __syncthreads();
    for (int b = threadIdx.x; b < B_; b += blockDim.x)
        atomicAdd(&s[label[b]], visible[b * V_ + v]);
    __syncthreads();
    for (int k = threadIdx.x; k < K_; k += blockDim.x)
        denom[k * V_ + v] = s[k];
}

// ---------------------------------------------------------------------------
// Kernel 3: update[k, v*C+c] = sum_b onehot[k,b] * vertices[b,v,c]*visible[b,v]
// via v_wmma_f32_16x16x32_f16 with exact hi/lo f16 split of the f32 operand.
// One wave per 16-column strip of the flattened (v,c) axis; 4 waves/block,
// all 4 strips of a block share the same v (visible cached in LDS).
// ---------------------------------------------------------------------------
__global__ void __launch_bounds__(128)
NM_segsum_wmma(const float* __restrict__ vertices,
               const float* __restrict__ visible,
               const v16h* __restrict__ Aop,
               float* __restrict__ update /* = d_out scratch */) {
    const int wave  = threadIdx.x >> 5;
    const int lane  = threadIdx.x & 31;
    const int strip = blockIdx.x * 4 + wave;         // 0..8191
    const int v     = strip >> 3;                    // C=128 -> 8 strips per v
    const int c0    = (strip & 7) << 4;
    const int n     = lane & 15;                     // B-matrix column (N)
    const int bb    = (lane < 16) ? 0 : 16;          // contiguous K-half, matches Aop

    // visible[., v] shared by all 4 waves of this block (v == blockIdx.x>>1)
    __shared__ float vis_s[B_];
    for (int i = threadIdx.x; i < B_; i += 128) vis_s[i] = visible[i * V_ + v];
    __syncthreads();

    v8f acc[KT] = {};   // 7 tiles x 8 VGPR f32 accumulators

    const float* colbase = vertices + (size_t)v * C_ + c0 + n;
    for (int bc = 0; bc < BC; ++bc) {
        const int b0 = bc * 32 + bb;
        if (bc + 1 < BC)   // speculative prefetch of next chunk's column data
            __builtin_prefetch(colbase + (size_t)(b0 + 32) * (V_ * C_), 0, 0);

        // Build B operand (32x16 f16, K=batch): lane holds 16 consecutive
        // batches b0..b0+15 of its column. Split f32 -> hi + lo f16 exactly.
        v16h bhi, blo;
#pragma unroll
        for (int h = 0; h < 16; ++h) {
            const int b = b0 + h;
            const float w  = colbase[(size_t)b * (V_ * C_)] * vis_s[b];
            const _Float16 hi = (_Float16)w;
            const _Float16 lo = (_Float16)(w - (float)hi);
            bhi[h] = hi;
            blo[h] = lo;
        }
#pragma unroll
        for (int t = 0; t < KT; ++t) {
            const v16h a = Aop[(t * BC + bc) * 32 + lane];
            acc[t] = __builtin_amdgcn_wmma_f32_16x16x32_f16(
                         false, a, false, bhi, (short)0, acc[t], false, false);
            acc[t] = __builtin_amdgcn_wmma_f32_16x16x32_f16(
                         false, a, false, blo, (short)0, acc[t], false, false);
        }
    }

    // C/D layout: VGPR r -> rows {r, r+8}; lanes 0-15 = N, 16-31 = N (M+8)
#pragma unroll
    for (int t = 0; t < KT; ++t)
#pragma unroll
        for (int r = 0; r < 8; ++r) {
            const int k = t * 16 + r + ((lane < 16) ? 0 : 8);
            if (k < K_)
                update[((size_t)k * V_ + v) * C_ + c0 + n] = acc[t][r];
        }
}

// ---------------------------------------------------------------------------
// Kernel 4: mean + normalize + EMA + normalize, one block per (k,v) row.
// ---------------------------------------------------------------------------
__device__ __forceinline__ float NM_block_sum(float x) {
    __shared__ float red[4];
#pragma unroll
    for (int off = 16; off > 0; off >>= 1) x += __shfl_xor(x, off, 32);
    const int w = threadIdx.x >> 5, l = threadIdx.x & 31;
    if (l == 0) red[w] = x;
    __syncthreads();
    const float tot = red[0] + red[1] + red[2] + red[3];
    __syncthreads();   // allow safe reuse of red[] on next call
    return tot;
}

__global__ void __launch_bounds__(C_)
NM_finalize(const float* __restrict__ weight,
            const float* __restrict__ denom,
            float* __restrict__ out) {
    const int kv  = blockIdx.x;                 // k*V + v
    const size_t off = (size_t)kv * C_ + threadIdx.x;

    float u = out[off] / fmaxf(denom[kv], 1.0f);
    const float n1 = sqrtf(NM_block_sum(u * u));
    const float un = u / fmaxf(n1, 1e-12f);

    const float nw = 0.9f * weight[off] + 0.1f * un;
    const float n2 = sqrtf(NM_block_sum(nw * nw));
    out[off] = nw / fmaxf(n2, 1e-12f);
}

// ---------------------------------------------------------------------------
extern "C" void kernel_launch(void* const* d_in, const int* in_sizes, int n_in,
                              void* d_out, int out_size, void* d_ws, size_t ws_size,
                              hipStream_t stream) {
    (void)in_sizes; (void)n_in; (void)out_size; (void)ws_size;
    const float* vertices = (const float*)d_in[0];   // (B,V,C) f32
    const float* visible  = (const float*)d_in[1];   // (B,V)   f32
    const int*   label    = (const int*)  d_in[2];   // (B,)    i32
    const float* weight   = (const float*)d_in[3];   // (K,V,C) f32
    float* out = (float*)d_out;                      // (K,V,C) f32

    char*  ws    = (char*)d_ws;
    v16h*  Aop   = (v16h*)ws;                        // 7*8*32*32B = 57,344 B (56 KB used: KT*BC*32*32)
    float* denom = (float*)(ws + 131072);            // K*V*4 = 409,600 B

    NM_build_a<<<KT * BC, 32, 0, stream>>>(label, Aop);
    NM_denom<<<V_, 128, 0, stream>>>(visible, label, denom);
    NM_segsum_wmma<<<(V_ * C_ / 16) / 4, 128, 0, stream>>>(vertices, visible, Aop, out);
    NM_finalize<<<K_ * V_, C_, 0, stream>>>(weight, denom, out);
}
